// MultiHeadAttention_15831249453371
// MI455X (gfx1250) — matrix operand order repaired and checked by reading, not compiled, not run on hardware
//
#include <hip/hip_runtime.h>
#include <hip/hip_bf16.h>

#define BDIM 2
#define LDIM 2048
#define DDIM 1024
#define HDIM 16
#define HD   64

typedef __attribute__((ext_vector_type(16))) __bf16 v16bf;
typedef __attribute__((ext_vector_type(8)))  __bf16 v8bf;
typedef __attribute__((ext_vector_type(8)))  float  v8f;
typedef unsigned int uint32x4 __attribute__((ext_vector_type(4)));
typedef int          int32x8  __attribute__((ext_vector_type(8)));
typedef int          int32x4  __attribute__((ext_vector_type(4)));

#if defined(__has_builtin)
#if __has_builtin(__builtin_amdgcn_tensor_load_to_lds) && __has_builtin(__builtin_amdgcn_s_wait_tensorcnt)
#define HAVE_TDM 1
#endif
#endif
#ifndef HAVE_TDM
#define HAVE_TDM 0
#endif

static __device__ inline v8f wmma_bf16(v16bf a, v16bf b, v8f c) {
  // (neg_a, A, neg_b, B, c_mod, C, reuse_a, reuse_b)
  return __builtin_amdgcn_wmma_f32_16x16x32_bf16(false, a, false, b, (short)0, c, false, false);
}

static __device__ inline v16bf mk16(v8bf lo, v8bf hi) {
  v16bf r;
#pragma unroll
  for (int i = 0; i < 8; ++i) { r[i] = lo[i]; r[8 + i] = hi[i]; }
  return r;
}

#if HAVE_TDM
// 2-D TDM tile load: til1 rows x til0 elems (2-byte data), row stride = stride
// elems, into contiguous LDS at lds_off. Descriptor per CDNA5 ISA sec 8.3/8.4.
static __device__ __attribute__((always_inline)) void tdm_load_2d(
    const void* gsrc, unsigned lds_off, int tdim0, int tdim1,
    int til0, int til1, int stride) {
  const unsigned long long ga = (unsigned long long)gsrc;
  uint32x4 g0;
  g0[0] = 1u;                                            // count=1, user mode
  g0[1] = lds_off;                                       // lds_addr (bytes)
  g0[2] = (unsigned)ga;                                  // global_addr[31:0]
  g0[3] = ((unsigned)(ga >> 32) & 0x1ffffffu) | (2u << 30);  // addr[56:32]|type=2
  int32x8 g1;
  g1[0] = (int)(1u << 16);                               // data_size=1 (2 bytes)
  g1[1] = (int)(((unsigned)tdim0 & 0xffffu) << 16);      // tensor_dim0[15:0]
  g1[2] = (int)(((unsigned)tdim0 >> 16) | (((unsigned)tdim1 & 0xffffu) << 16));
  g1[3] = (int)(((unsigned)tdim1 >> 16) | (((unsigned)til0 & 0xffffu) << 16));
  g1[4] = (int)((unsigned)til1 & 0xffffu);               // tile_dim1, tile_dim2=0
  g1[5] = stride;                                        // dim0_stride[31:0]
  g1[6] = 0;
  g1[7] = 0;
  const int32x4 z4 = {0, 0, 0, 0};
#if __clang_major__ >= 23
  const int32x8 z8 = {0, 0, 0, 0, 0, 0, 0, 0};
  __builtin_amdgcn_tensor_load_to_lds(g0, g1, z4, z4, z8, 0);
#else
  __builtin_amdgcn_tensor_load_to_lds(g0, g1, z4, z4, 0);
#endif
}
#endif

// ---------------------------------------------------------------------------
// fp32 -> bf16 bulk conversion (memory bound, runs once per tensor)
// ---------------------------------------------------------------------------
__global__ __launch_bounds__(256) void cvt_bf16_kernel(
    const float* __restrict__ in, __bf16* __restrict__ out, int n) {
  const int i = (blockIdx.x * 256 + threadIdx.x) * 8;
  if (i >= n) return;
  const float4 a = *(const float4*)(in + i);
  const float4 b = *(const float4*)(in + i + 4);
  v8bf o;
  o[0] = (__bf16)a.x; o[1] = (__bf16)a.y; o[2] = (__bf16)a.z; o[3] = (__bf16)a.w;
  o[4] = (__bf16)b.x; o[5] = (__bf16)b.y; o[6] = (__bf16)b.z; o[7] = (__bf16)b.w;
  *(v8bf*)(out + i) = o;
}

// ---------------------------------------------------------------------------
// Unified GEMM: Y = X @ W.T + b, bf16 operands, fp32 accumulate.
// Block = 4 waves; wave tile = 32 rows x 64 cols (8 WMMAs / K-step).
// W tile (64x32) staged through LDS by the Tensor Data Mover, double-buffered
// and overlapped with compute via TENSORcnt.
// mode 0: bf16 out [B,H,L,HD] scaled   mode 1: bf16 out [B,H,HD,L] (V^T)
// mode 2: fp32 out [B*L, D]
// ---------------------------------------------------------------------------
__global__ __launch_bounds__(128) void gemm_kernel(
    const __bf16* __restrict__ X, const __bf16* __restrict__ W,
    const float* __restrict__ bias, void* __restrict__ outp,
    float scale, int mode) {
  const int lane = threadIdx.x & 31;
  const int wave = threadIdx.x >> 5;
  const int half = lane >> 4;
  const int l16  = lane & 15;
  const int row_base = blockIdx.x * 128 + wave * 32;
  const int col_base = blockIdx.y * 64;

  v8f acc[2][4] = {};

#if HAVE_TDM
  __shared__ __attribute__((aligned(16))) __bf16 Bt[2][64 * 32];
  const __bf16* wtile = W + (long)col_base * DDIM;       // uniform
  if (wave == 0)
    tdm_load_2d(wtile, (unsigned)(unsigned long long)&Bt[0][0],
                DDIM, DDIM, 32, 64, DDIM);
  int cur = 0;
#endif

#pragma unroll 1
  for (int k = 0; k < DDIM; k += 32) {
#if HAVE_TDM
    __builtin_amdgcn_s_wait_tensorcnt(0);  // no-op for waves that issued none
    __syncthreads();   // tile ready; all waves past last iteration's reads
    if (wave == 0 && k + 32 < DDIM)
      tdm_load_2d(wtile + (k + 32), (unsigned)(unsigned long long)&Bt[cur ^ 1][0],
                  DDIM, DDIM, 32, 64, DDIM);
#endif
    // A fragments (global, long latency) issued first
    v16bf a[2];
#pragma unroll
    for (int r = 0; r < 2; ++r) {
      const v8bf* ap = (const v8bf*)(X + (long)(row_base + 16 * r + l16) * DDIM + k);
      a[r] = mk16(ap[half], ap[2 + half]);          // K chunks 8*half, 16+8*half
    }
    // all four B fragments up front -> one load clause, no per-pair dscnt stalls
    v16bf bm[4];
#pragma unroll
    for (int t = 0; t < 4; ++t) {
#if HAVE_TDM
      const v8bf* wp = (const v8bf*)(&Bt[cur][(16 * t + l16) * 32]);
#else
      const v8bf* wp = (const v8bf*)(W + (long)(col_base + 16 * t + l16) * DDIM + k);
#endif
      bm[t] = mk16(wp[half], wp[2 + half]);         // K chunks 8*half, 16+8*half
    }
#pragma unroll
    for (int t = 0; t < 4; ++t) {
      acc[0][t] = wmma_bf16(a[0], bm[t], acc[0][t]);
      acc[1][t] = wmma_bf16(a[1], bm[t], acc[1][t]);
    }
#if HAVE_TDM
    cur ^= 1;
#endif
  }

#pragma unroll
  for (int t = 0; t < 4; ++t) {
    const int e  = col_base + 16 * t + l16;
    const float bv = bias[e];
    const int hh = e >> 6, hd = e & (HD - 1);
#pragma unroll
    for (int r = 0; r < 2; ++r) {
#pragma unroll
      for (int v = 0; v < 8; ++v) {
        const int n  = row_base + 16 * r + v + 8 * half;  // C row = vgpr + 8*half
        const int bb = n >> 11, ll = n & (LDIM - 1);
        const float val = (acc[r][t][v] + bv) * scale;
        if (mode == 0) {
          ((__bf16*)outp)[((long)(bb * HDIM + hh) * LDIM + ll) * HD + hd] = (__bf16)val;
        } else if (mode == 1) {
          ((__bf16*)outp)[((long)(bb * HDIM + hh) * HD + hd) * LDIM + ll] = (__bf16)val;
        } else {
          ((float*)outp)[(long)n * DDIM + e] = val;
        }
      }
    }
  }
}

// ---------------------------------------------------------------------------
// Flash attention: one wave per (b,h,16-query tile), 64-key chunks
// (16 WMMAs per chunk). Q pre-scaled by 1/sqrt(HD). Softmax row stats are
// indexed by C-fragment vgpr slot (row = v + 8*half), replicated across the
// 16 lanes of each half; butterfly reductions stay inside each half.
// V fragments are issued after the K loads (LOADcnt completes in order) and
// before the softmax, so the softmax VALU block hides their latency.
// ---------------------------------------------------------------------------
__global__ __launch_bounds__(32) void attn_kernel(
    const __bf16* __restrict__ Qb, const __bf16* __restrict__ Kb,
    const __bf16* __restrict__ Vt, const unsigned char* __restrict__ mask,
    __bf16* __restrict__ AO) {
  const int lane = threadIdx.x & 31;
  const int half = lane >> 4;
  const int l16  = lane & 15;
  const int nqt  = LDIM / 16;
  const int qt = blockIdx.x % nqt;
  const int bh = blockIdx.x / nqt;
  const int b  = bh / HDIM;
  const int h  = bh % HDIM;

  const __bf16* Qp = Qb + (long)bh * LDIM * HD;
  const __bf16* Kp = Kb + (long)bh * LDIM * HD;
  const __bf16* Vp = Vt + (long)bh * HD * LDIM;
  const unsigned char* mp = mask + (long)b * LDIM;

  const int qrow = qt * 16 + l16;
  const v8bf* qv = (const v8bf*)(Qp + (long)qrow * HD);
  const v16bf Qa0 = mk16(qv[half],     qv[2 + half]);   // hd 0..31
  const v16bf Qa1 = mk16(qv[4 + half], qv[6 + half]);   // hd 32..63

  float mrow[8], lrow[8];
#pragma unroll
  for (int v = 0; v < 8; ++v) { mrow[v] = -1e30f; lrow[v] = 0.f; }
  v8f O[4] = {};

  __shared__ __attribute__((aligned(16))) __bf16 Pl[16][64];

#pragma unroll 1
  for (int kc = 0; kc < LDIM / 64; ++kc) {
    const int k0 = kc * 64;

    // scores (K loads + 8 WMMAs)
    v8f S[4];
    bool msk[4];
#pragma unroll
    for (int t = 0; t < 4; ++t) {
      const v8bf* kp = (const v8bf*)(Kp + (long)(k0 + 16 * t + l16) * HD);
      const v16bf Kf0 = mk16(kp[half],     kp[2 + half]);   // hd 0..31
      const v16bf Kf1 = mk16(kp[4 + half], kp[6 + half]);   // hd 32..63
      v8f s = {};
      s = wmma_bf16(Qa0, Kf0, s);
      s = wmma_bf16(Qa1, Kf1, s);
      S[t] = s;
      msk[t] = mp[k0 + 16 * t + l16] != 0;    // column key of tile t
    }

    // V fragments: issue now, consume after softmax (latency hidden)
    v16bf Vf0[4], Vf1[4];
#pragma unroll
    for (int t = 0; t < 4; ++t) {
      const v8bf* vp = (const v8bf*)(Vp + (long)(16 * t + l16) * LDIM + k0 + 16 * half);
      Vf0[t] = mk16(vp[0], vp[1]);
      Vf1[t] = mk16(vp[4], vp[5]);   // +32 keys
    }

    // online softmax over the 16x64 chunk
    float p[4][8], corr[8];
#pragma unroll
    for (int v = 0; v < 8; ++v) {
      float sv[4];
#pragma unroll
      for (int t = 0; t < 4; ++t) sv[t] = msk[t] ? -INFINITY : S[t][v];
      float mx = fmaxf(fmaxf(sv[0], sv[1]), fmaxf(sv[2], sv[3]));
#pragma unroll
      for (int s = 1; s < 16; s <<= 1) mx = fmaxf(mx, __shfl_xor(mx, s, 32));
      const float mnew = fmaxf(mrow[v], mx);
      corr[v] = __expf(mrow[v] - mnew);
      float cs = 0.f;
#pragma unroll
      for (int t = 0; t < 4; ++t) { p[t][v] = __expf(sv[t] - mnew); cs += p[t][v]; }
#pragma unroll
      for (int s = 1; s < 16; s <<= 1) cs += __shfl_xor(cs, s, 32);
      lrow[v] = lrow[v] * corr[v] + cs;
      mrow[v] = mnew;
    }
#pragma unroll
    for (int t = 0; t < 4; ++t)
#pragma unroll
      for (int v = 0; v < 8; ++v) O[t][v] *= corr[v];

    // C-layout fp32 -> A-layout bf16 via LDS bounce
#pragma unroll
    for (int v = 0; v < 8; ++v)
#pragma unroll
      for (int t = 0; t < 4; ++t)
        Pl[v + 8 * half][16 * t + l16] = (__bf16)p[t][v];
    __syncthreads();
    const v8bf* pp = (const v8bf*)(&Pl[l16][0]);
    const v16bf Pa0 = mk16(pp[2 * half],     pp[2 * half + 1]);   // keys k0+0..31
    const v16bf Pa1 = mk16(pp[4 + 2 * half], pp[5 + 2 * half]);   // keys k0+32..63
    __syncthreads();

    // P·V
#pragma unroll
    for (int t = 0; t < 4; ++t) {
      O[t] = wmma_bf16(Pa0, Vf0[t], O[t]);
      O[t] = wmma_bf16(Pa1, Vf1[t], O[t]);
    }
  }

#pragma unroll
  for (int t = 0; t < 4; ++t) {
#pragma unroll
    for (int v = 0; v < 8; ++v) {
      const float val = O[t][v] / lrow[v];
      const int ll = qt * 16 + v + 8 * half;
      const int e  = h * HD + 16 * t + l16;
      AO[(long)(b * LDIM + ll) * DDIM + e] = (__bf16)val;
    }
  }
}

extern "C" void kernel_launch(void* const* d_in, const int* in_sizes, int n_in,
                              void* d_out, int out_size, void* d_ws, size_t ws_size,
                              hipStream_t stream) {
  (void)in_sizes; (void)n_in; (void)out_size; (void)ws_size;
  const float* q  = (const float*)d_in[0];
  const float* k  = (const float*)d_in[1];
  const float* v  = (const float*)d_in[2];
  const unsigned char* mask = (const unsigned char*)d_in[3];
  const float* Wq = (const float*)d_in[4];
  const float* bq = (const float*)d_in[5];
  const float* Wk = (const float*)d_in[6];
  const float* bk = (const float*)d_in[7];
  const float* Wv = (const float*)d_in[8];
  const float* bv = (const float*)d_in[9];
  const float* Wo = (const float*)d_in[10];
  const float* bo = (const float*)d_in[11];
  float* out = (float*)d_out;

  const size_t act = (size_t)BDIM * LDIM * DDIM;    // 4M elems
  const size_t wgt = (size_t)DDIM * DDIM;           // 1M elems
  __bf16* Qb = (__bf16*)d_ws;        // 8 MB
  __bf16* Kb = Qb + act;             // 8 MB
  __bf16* Vt = Kb + act;             // 8 MB
  __bf16* AO = Vt + act;             // 8 MB
  __bf16* Xb = AO + act;             // 8 MB (reused for q, k, v)
  __bf16* Wb = Xb + act;             // 2 MB (reused for Wq..Wo) -> 42 MB total

  const int actBlk = (int)(act / 8 / 256);   // 2048
  const int wgtBlk = (int)(wgt / 8 / 256);   // 512
  dim3 gg(BDIM * LDIM / 128, DDIM / 64);     // 32 x 16
  const float scale = 0.125f;                // 1/sqrt(HD)

  cvt_bf16_kernel<<<actBlk, 256, 0, stream>>>(q, Xb, (int)act);
  cvt_bf16_kernel<<<wgtBlk, 256, 0, stream>>>(Wq, Wb, (int)wgt);
  gemm_kernel<<<gg, 128, 0, stream>>>(Xb, Wb, bq, Qb, scale, 0);

  cvt_bf16_kernel<<<actBlk, 256, 0, stream>>>(k, Xb, (int)act);
  cvt_bf16_kernel<<<wgtBlk, 256, 0, stream>>>(Wk, Wb, (int)wgt);
  gemm_kernel<<<gg, 128, 0, stream>>>(Xb, Wb, bk, Kb, 1.0f, 0);

  cvt_bf16_kernel<<<actBlk, 256, 0, stream>>>(v, Xb, (int)act);
  cvt_bf16_kernel<<<wgtBlk, 256, 0, stream>>>(Wv, Wb, (int)wgt);
  gemm_kernel<<<gg, 128, 0, stream>>>(Xb, Wb, bv, Vt, 1.0f, 1);

  attn_kernel<<<BDIM * HDIM * (LDIM / 16), 32, 0, stream>>>(Qb, Kb, Vt, mask, AO);

  cvt_bf16_kernel<<<wgtBlk, 256, 0, stream>>>(Wo, Wb, (int)wgt);
  gemm_kernel<<<gg, 128, 0, stream>>>(AO, Wb, bo, out, 1.0f, 2);
}
